// CausalSelfAttention_82351702934206
// MI455X (gfx1250) — matrix operand order repaired and checked by reading, not compiled, not run on hardware
//
#include <hip/hip_runtime.h>

// Problem sizes (fixed by the reference)
#define BB   8
#define TB   2048
#define CC   1024
#define MTOT (BB * TB)   // 16384 rows of x
#define N3C  (3 * CC)    // 3072

// LDS B-tile geometry: 64 rows x 32 bf16, padded to 80B row pitch (bank spread)
#define LDSROW  40               // ushorts per row (32 data + 8 pad)
#define STAGEE  (64 * LDSROW)    // ushorts per stage
#define STAGEB  (STAGEE * 2)     // bytes per stage

typedef __attribute__((ext_vector_type(16))) __bf16 v16bf;
typedef __attribute__((ext_vector_type(8)))  float  v8f;
typedef __attribute__((ext_vector_type(4)))  unsigned int v4u;
typedef __attribute__((ext_vector_type(8)))  int v8i_t;
typedef __attribute__((ext_vector_type(4)))  int v4i_t;

// ---------------------------------------------------------------- helpers ---

__device__ __forceinline__ unsigned short f2bf(float f) {
  union { float f; unsigned int u; } c; c.f = f;
  unsigned int u = c.u;
  unsigned int r = u + 0x7FFFu + ((u >> 16) & 1u);   // round-to-nearest-even
  return (unsigned short)(r >> 16);
}

// Low 32 bits of a generic pointer to __shared__ == hardware LDS byte address
__device__ __forceinline__ unsigned lds_off(const void* p) {
  return (unsigned)(unsigned long long)p;
}

// A-operand fragment, 16x32 bf16, source row-major [M,K] (K contiguous).
// lane L -> M = L&15, half = L>>4; VGPR i holds K-pair
//   k = (i<4 ? 0 : 16) + 8*half + 2*(i&3)
__device__ __forceinline__ v16bf load_a_frag(const unsigned short* A, int lda) {
  const int lane = threadIdx.x & 31;
  const int m = lane & 15, h = lane >> 4;
  union { v16bf v; unsigned int u[8]; } r;
  const unsigned short* row = A + m * lda + h * 8;
#pragma unroll
  for (int i = 0; i < 8; ++i) {
    const int k = ((i < 4) ? 0 : 16) + 2 * (i & 3);
    r.u[i] = *(const unsigned int*)(row + k);
  }
  return r.v;
}

// B-operand fragment from an LDS-staged tile (row pitch LDSROW ushorts).
// lane L -> N = L&15, half = L>>4; VGPR i holds K-pair
//   k = (i<4 ? 0 : 16) + 8*half + 2*(i&3)
__device__ __forceinline__ v16bf load_b_frag_lds(const unsigned short* ldsB) {
  const int lane = threadIdx.x & 31;
  const int n = lane & 15, h = lane >> 4;
  union { v16bf v; unsigned int u[8]; } r;
  const unsigned short* row = ldsB + n * LDSROW + h * 8;
#pragma unroll
  for (int i = 0; i < 8; ++i)
    r.u[i] = *(const unsigned int*)(row + ((i < 4) ? 0 : 16) + 2 * (i & 3));
  return r.v;
}

__device__ __forceinline__ v8f wmma_bf16(v16bf a, v16bf b, v8f c) {
  return __builtin_amdgcn_wmma_f32_16x16x32_bf16(
      /*neg_a=*/false, a, /*neg_b=*/false, b,
      /*c_mod=*/(short)0, c, /*reuse_a=*/false, /*reuse_b=*/false);
}

// ------------------------------------------------ Tensor Data Mover staging --

#if __has_builtin(__builtin_amdgcn_tensor_load_to_lds)
#define HAVE_TDM 1
// DMA a 64-row x 32-element bf16 tile (row stride ldb elements) into LDS at
// lds_byte_addr, with 16B padding after each 64B row (=> 80B LDS pitch).
__device__ __forceinline__ void tdm_load_b64x32(const unsigned short* gsrc, int ldb,
                                                unsigned lds_byte_addr) {
  unsigned long long ga = (unsigned long long)(const void*)gsrc;
  v4u g0;
  g0[0] = 1u;                                            // count=1, user D#
  g0[1] = lds_byte_addr;                                 // lds_addr
  g0[2] = (unsigned)ga;                                  // global_addr[31:0]
  g0[3] = (unsigned)((ga >> 32) & 0x01FFFFFFull) | (2u << 30);  // addr[56:32], type=2
  v8i_t g1;
  // data_size=2B | pad_enable | pad_interval=16dw(64B) | pad_amount=4dw(16B)
  g1[0] = (int)((1u << 16) | (1u << 20) | (3u << 22) | (3u << 25));
  g1[1] = (int)(32u << 16);   // tensor_dim0 = 32  (bits 79:48)
  g1[2] = (int)(64u << 16);   // tensor_dim1 = 64  (bits 111:80)
  g1[3] = (int)(32u << 16);   // tile_dim0  = 32  (bits 127:112)
  g1[4] = 64;                 // tile_dim1 = 64, tile_dim2 = 0
  g1[5] = ldb;                // tensor_dim0_stride[31:0]
  g1[6] = 0;                  // stride hi | tensor_dim1_stride lo
  g1[7] = 0;
  v4i_t gz = {0, 0, 0, 0};
#if __clang_major__ >= 23
  v8i_t gz8 = {0, 0, 0, 0, 0, 0, 0, 0};
  __builtin_amdgcn_tensor_load_to_lds(g0, g1, gz, gz, gz8, 0);
#else
  __builtin_amdgcn_tensor_load_to_lds(g0, g1, gz, gz, 0);
#endif
}
#else
#define HAVE_TDM 0
#endif

// Cooperative fallback copy of one 64x32 tile into a padded LDS stage.
__device__ __forceinline__ void copy_b64x32(const unsigned short* gsrc, int ldb,
                                            unsigned short* dst) {
  const int row = threadIdx.x >> 2, ch = threadIdx.x & 3;   // 256 thr * 16B = 8KB... 4KB data
  *(uint4*)(dst + row * LDSROW + ch * 8) =
      *(const uint4*)(gsrc + (size_t)row * ldb + ch * 8);
}

// ------------------------------------------------------- block GEMM core ----
// Block: 8 waves; wave w owns rows [m0b + 32w, +32). All waves share the
// 64-column B tile, staged in LDS (double-buffered, TDM-prefetched).
// Must be called by all 256 threads (contains barriers).
__device__ __forceinline__ void gemm_block(const unsigned short* A, int lda,
                                           const unsigned short* BTtile, int ldb,
                                           int K, bool active,
                                           unsigned short* ldsB,  // [2*STAGEE]
                                           v8f acc[2][4]) {
  const int wave = threadIdx.x >> 5;
  int cur = 0;

  // prologue: stage 0
#if HAVE_TDM
  if (wave == 0) {
    tdm_load_b64x32(BTtile, ldb, lds_off(ldsB));
    __builtin_amdgcn_s_wait_tensorcnt(0);
  }
#else
  copy_b64x32(BTtile, ldb, ldsB);
#endif
  __syncthreads();

  for (int k = 0; k < K; k += 32) {
    const int nxt = k + 32;
    if (nxt < K) {     // prefetch next stage while computing this one
#if HAVE_TDM
      if (wave == 0)
        tdm_load_b64x32(BTtile + nxt, ldb, lds_off(ldsB + (cur ^ 1) * STAGEE));
#else
      copy_b64x32(BTtile + nxt, ldb, ldsB + (cur ^ 1) * STAGEE);
#endif
    }
    if (active) {
      v16bf a[2], b[4];
#pragma unroll
      for (int mi = 0; mi < 2; ++mi)
        a[mi] = load_a_frag(A + (size_t)(mi * 16) * lda + k, lda);
#pragma unroll
      for (int ni = 0; ni < 4; ++ni)
        b[ni] = load_b_frag_lds(ldsB + cur * STAGEE + ni * 16 * LDSROW);
#pragma unroll
      for (int mi = 0; mi < 2; ++mi)
#pragma unroll
        for (int ni = 0; ni < 4; ++ni)
          acc[mi][ni] = wmma_bf16(a[mi], b[ni], acc[mi][ni]);
    }
    if (nxt < K) {
#if HAVE_TDM
      if (wave == 0) __builtin_amdgcn_s_wait_tensorcnt(0);
#endif
      __syncthreads();
      cur ^= 1;
    }
  }
}

__device__ __forceinline__ void zero_acc(v8f acc[2][4]) {
#pragma unroll
  for (int mi = 0; mi < 2; ++mi)
#pragma unroll
    for (int ni = 0; ni < 4; ++ni)
      acc[mi][ni] = (v8f){0.f, 0.f, 0.f, 0.f, 0.f, 0.f, 0.f, 0.f};
}

// ----------------------------------------------------------------- kernels --

__global__ void cvt_kernel(const float* __restrict__ x,
                           unsigned short* __restrict__ xb, int n) {
  int i = blockIdx.x * 256 + threadIdx.x;
  if (i < n) xb[i] = f2bf(x[i]);
}

// w [K][N] fp32 -> wT [N][K] bf16 (output-coalesced)
__global__ void transpose_cvt_kernel(const float* __restrict__ w,
                                     unsigned short* __restrict__ wT,
                                     int K, int N) {
  int idx = blockIdx.x * 256 + threadIdx.x;
  if (idx >= K * N) return;
  int n = idx / K, k = idx - n * K;
  wT[idx] = f2bf(w[k * N + n]);
}

// qkv = xb @ wAT^T + b_attn ; route to q / k (row-major [B*T,C]) and vT [B,C,T]
__global__ __launch_bounds__(256)
void qkv_gemm_kernel(const unsigned short* __restrict__ xb,
                     const unsigned short* __restrict__ wAT,
                     const float* __restrict__ b_attn,
                     unsigned short* __restrict__ q,
                     unsigned short* __restrict__ kbuf,
                     unsigned short* __restrict__ vT) {
  __shared__ __align__(16) unsigned short ldsB[2 * STAGEE];
  const int wave = threadIdx.x >> 5;
  const int lane = threadIdx.x & 31;
  const int m0 = blockIdx.y * 256 + wave * 32;
  const int n0 = blockIdx.x * 64;

  v8f acc[2][4];
  zero_acc(acc);
  gemm_block(xb + (size_t)m0 * CC, CC, wAT + (size_t)n0 * CC, CC, CC, true, ldsB, acc);

  const int nl = lane & 15, h = lane >> 4;
#pragma unroll
  for (int mi = 0; mi < 2; ++mi)
#pragma unroll
    for (int ni = 0; ni < 4; ++ni) {
      const int n = n0 + ni * 16 + nl;
      const int seg = n >> 10;         // 0:q 1:k 2:v
      const int c = n & (CC - 1);
      const float bias = b_attn[n];
#pragma unroll
      for (int r = 0; r < 8; ++r) {
        const int m = m0 + mi * 16 + r + 8 * h;
        const unsigned short val = f2bf(acc[mi][ni][r] + bias);
        if (seg == 0)      q[(size_t)m * CC + c] = val;
        else if (seg == 1) kbuf[(size_t)m * CC + c] = val;
        else {
          const int b = m >> 11, t = m & (TB - 1);
          vT[((size_t)b * CC + c) * TB + t] = val;
        }
      }
    }
}

// S[t][s] = (q_b[t]·k_b[s]) / 32 , causal mask with -1e10 (one batch)
__global__ __launch_bounds__(256)
void scores_kernel(const unsigned short* __restrict__ qb,
                   const unsigned short* __restrict__ kb,
                   float* __restrict__ S) {
  __shared__ __align__(16) unsigned short ldsB[2 * STAGEE];
  const int wave = threadIdx.x >> 5;
  const int lane = threadIdx.x & 31;
  const int m0b = blockIdx.y * 256;
  const int m0 = m0b + wave * 32;                // t
  const int n0 = blockIdx.x * 64;                // s
  const int nl = lane & 15, h = lane >> 4;

  if (n0 > m0b + 255) {                          // whole block above diagonal
#pragma unroll
    for (int mi = 0; mi < 2; ++mi)
#pragma unroll
      for (int ni = 0; ni < 4; ++ni)
#pragma unroll
        for (int r = 0; r < 8; ++r) {
          const int m = m0 + mi * 16 + r + 8 * h;
          const int n = n0 + ni * 16 + nl;
          S[(size_t)m * TB + n] = -10000000000.0f;
        }
    return;
  }

  const bool active = (n0 <= m0 + 31);           // wave has >=1 unmasked element
  v8f acc[2][4];
  zero_acc(acc);
  gemm_block(qb + (size_t)m0 * CC, CC, kb + (size_t)n0 * CC, CC, CC, active, ldsB, acc);

#pragma unroll
  for (int mi = 0; mi < 2; ++mi)
#pragma unroll
    for (int ni = 0; ni < 4; ++ni)
#pragma unroll
      for (int r = 0; r < 8; ++r) {
        const int m = m0 + mi * 16 + r + 8 * h;
        const int n = n0 + ni * 16 + nl;
        float val = acc[mi][ni][r] * 0.03125f;   // 1/sqrt(1024)
        if (n > m) val = -10000000000.0f;
        S[(size_t)m * TB + n] = val;
      }
}

// fp32 row softmax over 2048 entries -> bf16 P (one batch, block per row)
__global__ __launch_bounds__(256)
void softmax_kernel(const float* __restrict__ S, unsigned short* __restrict__ P) {
  __shared__ float red[256];
  const int t = blockIdx.x, tid = threadIdx.x;
  const float* row = S + (size_t)t * TB;

  float mx = -3.0e38f;
  for (int s = tid; s < TB; s += 256) mx = fmaxf(mx, row[s]);
  red[tid] = mx; __syncthreads();
  for (int off = 128; off > 0; off >>= 1) {
    if (tid < off) red[tid] = fmaxf(red[tid], red[tid + off]);
    __syncthreads();
  }
  mx = red[0]; __syncthreads();

  float sum = 0.f;
  for (int s = tid; s < TB; s += 256) sum += __expf(row[s] - mx);
  red[tid] = sum; __syncthreads();
  for (int off = 128; off > 0; off >>= 1) {
    if (tid < off) red[tid] += red[tid + off];
    __syncthreads();
  }
  const float inv = 1.0f / red[0];

  unsigned short* prow = P + (size_t)t * TB;
  for (int s = tid; s < TB; s += 256)
    prow[s] = f2bf(__expf(row[s] - mx) * inv);
}

// O_b = P @ v_b (one batch); block-uniform causal K truncation (P==0 beyond t)
__global__ __launch_bounds__(256)
void pv_kernel(const unsigned short* __restrict__ P,
               const unsigned short* __restrict__ vTb,
               unsigned short* __restrict__ obb) {
  __shared__ __align__(16) unsigned short ldsB[2 * STAGEE];
  const int wave = threadIdx.x >> 5;
  const int lane = threadIdx.x & 31;
  const int m0b = blockIdx.y * 256;
  const int m0 = m0b + wave * 32;                // t
  const int n0 = blockIdx.x * 64;                // c
  const int Keff = m0b + 256;                    // covers all waves' rows

  v8f acc[2][4];
  zero_acc(acc);
  gemm_block(P + (size_t)m0 * TB, TB, vTb + (size_t)n0 * TB, TB, Keff, true, ldsB, acc);

  const int nl = lane & 15, h = lane >> 4;
#pragma unroll
  for (int mi = 0; mi < 2; ++mi)
#pragma unroll
    for (int ni = 0; ni < 4; ++ni)
#pragma unroll
      for (int r = 0; r < 8; ++r) {
        const int m = m0 + mi * 16 + r + 8 * h;
        const int n = n0 + ni * 16 + nl;
        obb[(size_t)m * CC + n] = f2bf(acc[mi][ni][r]);
      }
}

// out = ob @ wPT^T + b_proj (fp32 output)
__global__ __launch_bounds__(256)
void proj_kernel(const unsigned short* __restrict__ ob,
                 const unsigned short* __restrict__ wPT,
                 const float* __restrict__ b_proj,
                 float* __restrict__ out) {
  __shared__ __align__(16) unsigned short ldsB[2 * STAGEE];
  const int wave = threadIdx.x >> 5;
  const int lane = threadIdx.x & 31;
  const int m0 = blockIdx.y * 256 + wave * 32;
  const int n0 = blockIdx.x * 64;

  v8f acc[2][4];
  zero_acc(acc);
  gemm_block(ob + (size_t)m0 * CC, CC, wPT + (size_t)n0 * CC, CC, CC, true, ldsB, acc);

  const int nl = lane & 15, h = lane >> 4;
#pragma unroll
  for (int mi = 0; mi < 2; ++mi)
#pragma unroll
    for (int ni = 0; ni < 4; ++ni) {
      const int n = n0 + ni * 16 + nl;
      const float bias = b_proj[n];
#pragma unroll
      for (int r = 0; r < 8; ++r) {
        const int m = m0 + mi * 16 + r + 8 * h;
        out[(size_t)m * CC + n] = acc[mi][ni][r] + bias;
      }
    }
}

// ------------------------------------------------------------------ launch --

extern "C" void kernel_launch(void* const* d_in, const int* in_sizes, int n_in,
                              void* d_out, int out_size, void* d_ws, size_t ws_size,
                              hipStream_t stream) {
  (void)in_sizes; (void)n_in; (void)out_size; (void)ws_size;

  const float* x      = (const float*)d_in[0];
  const float* w_attn = (const float*)d_in[1];
  const float* b_attn = (const float*)d_in[2];
  const float* w_proj = (const float*)d_in[3];
  const float* b_proj = (const float*)d_in[4];
  float* out = (float*)d_out;

  char* ws = (char*)d_ws;
  size_t off = 0;
  auto walloc = [&](size_t bytes) -> char* {
    char* p = ws + off;
    off += (bytes + 255) & ~(size_t)255;
    return p;
  };

  unsigned short* xb  = (unsigned short*)walloc((size_t)MTOT * CC * 2);  // x bf16
  unsigned short* wAT = (unsigned short*)walloc((size_t)N3C * CC * 2);   // w_attn^T bf16
  unsigned short* wPT = (unsigned short*)walloc((size_t)CC * CC * 2);    // w_proj^T bf16
  unsigned short* q   = (unsigned short*)walloc((size_t)MTOT * CC * 2);
  unsigned short* kb  = (unsigned short*)walloc((size_t)MTOT * CC * 2);
  unsigned short* vT  = (unsigned short*)walloc((size_t)MTOT * CC * 2);  // [B,C,T]
  float*          S   = (float*)walloc((size_t)TB * TB * 4);             // per-batch
  unsigned short* P   = (unsigned short*)walloc((size_t)TB * TB * 2);    // per-batch
  unsigned short* ob  = (unsigned short*)walloc((size_t)MTOT * CC * 2);

  dim3 blk(256);

  cvt_kernel<<<(MTOT * CC) / 256, blk, 0, stream>>>(x, xb, MTOT * CC);
  transpose_cvt_kernel<<<(CC * N3C + 255) / 256, blk, 0, stream>>>(w_attn, wAT, CC, N3C);
  transpose_cvt_kernel<<<(CC * CC + 255) / 256, blk, 0, stream>>>(w_proj, wPT, CC, CC);

  qkv_gemm_kernel<<<dim3(N3C / 64, MTOT / 256), blk, 0, stream>>>(xb, wAT, b_attn, q, kb, vT);

  for (int b = 0; b < BB; ++b) {
    const size_t qoff = (size_t)b * TB * CC;
    scores_kernel<<<dim3(TB / 64, TB / 256), blk, 0, stream>>>(q + qoff, kb + qoff, S);
    softmax_kernel<<<dim3(TB), blk, 0, stream>>>(S, P);
    pv_kernel<<<dim3(CC / 64, TB / 256), blk, 0, stream>>>(P, vT + (size_t)b * CC * TB, ob + qoff);
  }

  proj_kernel<<<dim3(CC / 64, MTOT / 256), blk, 0, stream>>>(ob, wPT, b_proj, out);
}
